// MHSA_37821482008969
// MI455X (gfx1250) — matrix operand order repaired and checked by reading, not compiled, not run on hardware
//
#include <hip/hip_runtime.h>

#define HEADS 4
#define DQK 128
#define DV 128
#define L 1024
#define DIM 512
#define QSCALE 0.08838834764831845f

typedef __attribute__((ext_vector_type(16))) __bf16 v16bf;
typedef __attribute__((ext_vector_type(8)))  __bf16 v8bf;
typedef __attribute__((ext_vector_type(8)))  float  v8f;
typedef unsigned int u32x4 __attribute__((ext_vector_type(4)));
typedef int i32x8 __attribute__((ext_vector_type(8)));
typedef int i32x4 __attribute__((ext_vector_type(4)));

union BF16Frag { v16bf v; uint4 q[2]; };

__device__ __forceinline__ v8f wmma_bf16(v16bf a, v16bf b, v8f c) {
  // D = A(16x32 bf16) * B(32x16 bf16) + C(16x16 f32)
  return __builtin_amdgcn_wmma_f32_16x16x32_bf16(false, a, false, b, (short)0, c,
                                                 false, false);
}

// A fragment 16(M)x32(K), src row-major [m][k]. Per lane: m=lane&15,
// k in [half*8, half*8+8) and [16+half*8, 16+half*8+8)  -> 2x16B loads.
__device__ __forceinline__ v16bf load_frag_a(const __bf16* src, int ld) {
  int lane = threadIdx.x & 31;
  int m = lane & 15, lh = lane >> 4;
  const __bf16* p = src + m * ld + lh * 8;
  BF16Frag f;
  f.q[0] = *(const uint4*)p;
  f.q[1] = *(const uint4*)(p + 16);
  return f.v;
}

// B fragment 32(K)x16(N), src stored [n][k] row-major (k contiguous).
// Per lane: n=lane&15, k in [half*8, half*8+8) and [16+half*8, 16+half*8+8) -> 2x16B loads.
__device__ __forceinline__ v16bf load_frag_b(const __bf16* src, int ld) {
  int lane = threadIdx.x & 31;
  int n = lane & 15, lh = lane >> 4;
  const __bf16* p = src + n * ld + lh * 8;
  BF16Frag f;
  f.q[0] = *(const uint4*)p;
  f.q[1] = *(const uint4*)(p + 16);
  return f.v;
}

__device__ __forceinline__ uint4 pack8_bf(v8f a, float s) {
  union { v8bf v; uint4 q; } u;
#pragma unroll
  for (int i = 0; i < 8; i++) u.v[i] = (__bf16)(a[i] * s);
  return u.q;
}

// ---------------------------------------------------------------------------
// Tensor Data Mover: 2D tile load Global -> LDS (bf16 elements, data_size=2B).
// D# group0: [1:0]=count=1, [63:32]=lds_addr, [120:64]=global_addr,
//            [127:126]=type(2).
// D# group1: [17:16]=data_size(1=2B), [79:48]=tensor_dim0, [111:80]=tensor_dim1,
//            [127:112]=tile_dim0, [143:128]=tile_dim1, [207:160]=dim0_stride.
// Issue once per workgroup (wave 0); tracked with TENSORcnt.
// This toolchain uses the 6-arg builtin: (g0, g1, g2, g3, g4, cpol);
// unused trailing groups are zero-filled (2-D tensor).
// ---------------------------------------------------------------------------
__device__ __forceinline__ void tdm_load_2d(unsigned lds_addr, const void* gptr,
                                            unsigned tile_d0, unsigned tile_d1,
                                            unsigned tensor_d0, unsigned tensor_d1,
                                            unsigned long long stride0) {
  unsigned long long ga = (unsigned long long)(uintptr_t)gptr;
  u32x4 g0;
  g0[0] = 1u;                                                 // count=1
  g0[1] = lds_addr;                                           // LDS byte addr
  g0[2] = (unsigned)ga;                                       // global addr lo
  g0[3] = (unsigned)((ga >> 32) & 0x01ffffffu) | (2u << 30);  // addr hi | type=2
  i32x8 g1;
  g1[0] = (1 << 16);                                          // data_size = 2B
  g1[1] = (int)((tensor_d0 & 0xffffu) << 16);
  g1[2] = (int)((tensor_d0 >> 16) | ((tensor_d1 & 0xffffu) << 16));
  g1[3] = (int)((tensor_d1 >> 16) | ((tile_d0 & 0xffffu) << 16));
  g1[4] = (int)(tile_d1 & 0xffffu);                           // tile_dim2 = 0
  g1[5] = (int)(stride0 & 0xffffffffu);
  g1[6] = (int)((stride0 >> 32) & 0xffffu);                   // dim1_stride = 0
  g1[7] = 0;
  i32x4 z4 = {};
  i32x8 z8 = {};
  __builtin_amdgcn_tensor_load_to_lds(g0, g1, z4, z4, z8, 0);
}

// ---------------------------------------------------------------------------
// Projection: Y[o,l] = sum_c W[o,c] * X[b,c,l], o in [0,1536)
//   o<512   -> Qb[b,h,l,d]  (scaled by 1/sqrt(128))
//   o<1024  -> Kb[b,h,l,d]
//   else    -> Vt[b,h,d,l]  (transposed for P*V B-operand)
// ---------------------------------------------------------------------------
__global__ __launch_bounds__(256) void proj_kernel(
    const float* __restrict__ x, const float* __restrict__ w_qk,
    const float* __restrict__ w_v, __bf16* __restrict__ Qb,
    __bf16* __restrict__ Kb, __bf16* __restrict__ Vt) {
  __shared__ __bf16 sW[64 * 32];    // [o][c]
  __shared__ __bf16 sX[128 * 32];   // [l][c]  (transposed stage)
  const int tid = threadIdx.x;
  const int l0 = blockIdx.x * 128;
  const int o0 = blockIdx.y * 64;
  const int b  = blockIdx.z;
  const int lane = tid & 31, wv = tid >> 5;
  const int osub = wv & 3, lbase = (wv >> 2) * 64;
  v8f acc[4] = {};
  for (int c0 = 0; c0 < DIM; c0 += 32) {
    __syncthreads();
#pragma unroll
    for (int i = 0; i < 8; i++) {            // stage W tile 64x32
      int idx = tid + i * 256;
      int ol = idx >> 5, cl = idx & 31;
      int o = o0 + ol, c = c0 + cl;
      float wval = (o < 1024) ? w_qk[o * DIM + c] : w_v[(o - 1024) * DIM + c];
      sW[ol * 32 + cl] = (__bf16)wval;
    }
#pragma unroll
    for (int i = 0; i < 16; i++) {           // stage X^T tile 128x32
      int idx = tid + i * 256;
      int ll = idx >> 5, cl = idx & 31;
      sX[ll * 32 + cl] = (__bf16)x[((size_t)b * DIM + c0 + cl) * L + l0 + ll];
    }
    __syncthreads();
    v16bf a = load_frag_a(sW + osub * 16 * 32, 32);
#pragma unroll
    for (int t = 0; t < 4; t++) {
      v16bf bfr = load_frag_b(sX + (lbase + t * 16) * 32, 32);
      acc[t] = wmma_bf16(a, bfr, acc[t]);
    }
  }
  const int n = lane & 15, lh = lane >> 4;
#pragma unroll
  for (int t = 0; t < 4; t++) {
    int l = l0 + lbase + t * 16 + n;
    int ob = o0 + osub * 16 + 8 * lh;        // 8 consecutive o in this lane's VGPRs
    if (ob < 512) {
      int head = ob >> 7, d0 = ob & 127;
      *(uint4*)(Qb + ((size_t)(b * HEADS + head) * L + l) * DQK + d0) =
          pack8_bf(acc[t], QSCALE);
    } else if (ob < 1024) {
      int oo = ob - 512, head = oo >> 7, d0 = oo & 127;
      *(uint4*)(Kb + ((size_t)(b * HEADS + head) * L + l) * DQK + d0) =
          pack8_bf(acc[t], 1.0f);
    } else {
      int oo = ob - 1024, head = oo >> 7, d0 = oo & 127;
#pragma unroll
      for (int r = 0; r < 8; r++)
        Vt[((size_t)(b * HEADS + head) * DV + d0 + r) * L + l] = (__bf16)acc[t][r];
    }
  }
}

// ---------------------------------------------------------------------------
// Relative-position tables: RWL[b,h,l,j'] = q[l] . rel_width[j'],  j' in [0,63)
// (stored with stride 64; column 63 unused). Same for RHL with rel_height.
// ---------------------------------------------------------------------------
__global__ __launch_bounds__(256) void relproj_kernel(
    const __bf16* __restrict__ Qb, const float* __restrict__ rel_h,
    const float* __restrict__ rel_w, float* __restrict__ RWL,
    float* __restrict__ RHL) {
  __shared__ __bf16 sRW[64 * 128];   // [j'][d], row 63 zero-padded
  __shared__ __bf16 sRH[64 * 128];
  const int tid = threadIdx.x;
  const int l0 = blockIdx.x * 128;
  const int bh = blockIdx.y;
#pragma unroll
  for (int i = 0; i < 32; i++) {
    int idx = tid + i * 256;
    int j = idx >> 7, d = idx & 127;
    sRW[idx] = (j < 63) ? (__bf16)rel_w[j * DQK + d] : (__bf16)0.0f;
    sRH[idx] = (j < 63) ? (__bf16)rel_h[j * DQK + d] : (__bf16)0.0f;
  }
  __syncthreads();
  const int lane = tid & 31, wv = tid >> 5;
  const __bf16* qrow = Qb + ((size_t)bh * L + l0 + wv * 16) * DQK;
  v16bf aq[4];
#pragma unroll
  for (int kf = 0; kf < 4; kf++) aq[kf] = load_frag_a(qrow + kf * 32, DQK);
  v8f accW[4] = {};
  v8f accH[4] = {};
#pragma unroll
  for (int nt = 0; nt < 4; nt++) {
#pragma unroll
    for (int kf = 0; kf < 4; kf++) {
      v16bf bw = load_frag_b(sRW + (nt * 16) * DQK + kf * 32, DQK);
      accW[nt] = wmma_bf16(aq[kf], bw, accW[nt]);
      v16bf bh2 = load_frag_b(sRH + (nt * 16) * DQK + kf * 32, DQK);
      accH[nt] = wmma_bf16(aq[kf], bh2, accH[nt]);
    }
  }
  const int n = lane & 15, lh = lane >> 4;
#pragma unroll
  for (int nt = 0; nt < 4; nt++) {
#pragma unroll
    for (int r = 0; r < 8; r++) {
      int l = l0 + wv * 16 + r + 8 * lh;
      int jp = nt * 16 + n;
      RWL[((size_t)bh * L + l) * 64 + jp] = accW[nt][r];
      RHL[((size_t)bh * L + l) * 64 + jp] = accH[nt][r];
    }
  }
}

// ---------------------------------------------------------------------------
// Flash attention: 128 rows of one (b,h) per block; 8 waves * 16 rows.
// K/V tiles DMA'd into double-buffered LDS by the Tensor Data Mover (wave 0
// issues, TENSORcnt-synchronized), overlapping the next tile's DMA with
// compute on the current one. Rel logits added via LDS gathers:
//   REL[l=(x,y), m=(i,j)] = RWL[l, j-y+31] + RHL[l, i-x+31]
// ---------------------------------------------------------------------------
#define ATTN_LDS (65536 /*rel*/ + 32768 /*K x2*/ + 32768 /*V x2*/ + 16384 /*P*/)

__global__ __launch_bounds__(256) void attn_kernel(
    const __bf16* __restrict__ Qb, const __bf16* __restrict__ Kb,
    const __bf16* __restrict__ Vt, const float* __restrict__ RWL,
    const float* __restrict__ RHL, float* __restrict__ out) {
  extern __shared__ char smem[];
  float* sRelW = (float*)smem;                       // 128*64 f32
  float* sRelH = (float*)(smem + 32768);             // 128*64 f32
  __bf16* sKb0 = (__bf16*)(smem + 65536);            // 64*128 bf16 [y][d], buf 0
  __bf16* sKb1 = (__bf16*)(smem + 81920);            //                     buf 1
  __bf16* sVb0 = (__bf16*)(smem + 98304);            // 128*64 bf16 [d][y], buf 0
  __bf16* sVb1 = (__bf16*)(smem + 114688);           //                     buf 1
  __bf16* sP   = (__bf16*)(smem + 131072);           // 8 waves * 16*64 bf16
  const int tid = threadIdx.x;
  const int l0 = blockIdx.x * 128;
  const int bh = blockIdx.y;
  const int lane = tid & 31, wv = tid >> 5;
  const int n = lane & 15, lh = lane >> 4;

  {  // stage rel tables for this block's 128 rows
    const float4* srcW = (const float4*)(RWL + ((size_t)bh * L + l0) * 64);
    const float4* srcH = (const float4*)(RHL + ((size_t)bh * L + l0) * 64);
    float4* dW = (float4*)sRelW;
    float4* dH = (float4*)sRelH;
#pragma unroll
    for (int i = 0; i < 8; i++) {
      dW[tid + i * 256] = srcW[tid + i * 256];
      dH[tid + i * 256] = srcH[tid + i * 256];
    }
  }

  // wave 0 drives the TDM: K tile is 64*128 bf16 contiguous (1D), V tile is
  // 128 rows x 64 cols with row stride 1024 elements (2D).
  if (wv == 0) {
    tdm_load_2d((unsigned)(uintptr_t)sKb0, Kb + ((size_t)bh * L) * DQK,
                8192u, 0u, 8192u, 1u, 8192ull);
    tdm_load_2d((unsigned)(uintptr_t)sVb0, Vt + (size_t)bh * DV * L,
                64u, 128u, 1024u, 128u, 1024ull);
  }

  const __bf16* qrow = Qb + ((size_t)bh * L + l0 + wv * 16) * DQK;
  v16bf aq[4];
#pragma unroll
  for (int kf = 0; kf < 4; kf++) aq[kf] = load_frag_a(qrow + kf * 32, DQK);

  v8f acc[8] = {};
  float mrow[8], lrow[8];
#pragma unroll
  for (int r = 0; r < 8; r++) { mrow[r] = -1e30f; lrow[r] = 0.0f; }

  __bf16* sPw = sP + wv * (16 * 64);
  const int row0 = wv * 16 + 8 * lh;   // block-local row = row0 + r

  for (int cb = 0; cb < 16; cb++) {
    const int buf = cb & 1;
    __bf16* sK = buf ? sKb1 : sKb0;
    __bf16* sV = buf ? sVb1 : sVb0;
    if (wv == 0) __builtin_amdgcn_s_wait_tensorcnt(0);  // current buf DMA done
    __syncthreads();  // all waves done with previous tiles; data visible
    if (wv == 0 && cb < 15) {  // prefetch next tile into the other buffer
      __bf16* nK = buf ? sKb0 : sKb1;
      __bf16* nV = buf ? sVb0 : sVb1;
      tdm_load_2d((unsigned)(uintptr_t)nK,
                  Kb + ((size_t)bh * L + (cb + 1) * 64) * DQK,
                  8192u, 0u, 8192u, 1u, 8192ull);
      tdm_load_2d((unsigned)(uintptr_t)nV,
                  Vt + (size_t)bh * DV * L + (cb + 1) * 64,
                  64u, 128u, 1024u, 128u, 1024ull);
    }

    // S = Q * K^T  (4 col tiles x 4 k fragments)
    v8f s[4] = {};
#pragma unroll
    for (int t = 0; t < 4; t++) {
#pragma unroll
      for (int kf = 0; kf < 4; kf++) {
        v16bf bk = load_frag_b(sK + (t * 16) * DQK + kf * 32, DQK);
        s[t] = wmma_bf16(aq[kf], bk, s[t]);
      }
    }
    // add relative-position logits
#pragma unroll
    for (int t = 0; t < 4; t++) {
      int m = cb * 64 + t * 16 + n;
      int i2 = m >> 5, j2 = m & 31;
#pragma unroll
      for (int r = 0; r < 8; r++) {
        int rowl = row0 + r;
        int lg = l0 + rowl;
        int xx = lg >> 5, yy = lg & 31;
        s[t][r] += sRelW[rowl * 64 + (j2 - yy + 31)] +
                   sRelH[rowl * 64 + (i2 - xx + 31)];
      }
    }
    // online softmax (row stats replicated across the 16 lanes of a half)
    float alpha[8];
#pragma unroll
    for (int r = 0; r < 8; r++) {
      float rm = fmaxf(fmaxf(s[0][r], s[1][r]), fmaxf(s[2][r], s[3][r]));
      rm = fmaxf(rm, __shfl_xor(rm, 1));
      rm = fmaxf(rm, __shfl_xor(rm, 2));
      rm = fmaxf(rm, __shfl_xor(rm, 4));
      rm = fmaxf(rm, __shfl_xor(rm, 8));
      float mn = fmaxf(mrow[r], rm);
      alpha[r] = __expf(mrow[r] - mn);
      mrow[r] = mn;
      float ps = 0.0f;
#pragma unroll
      for (int t = 0; t < 4; t++) { s[t][r] = __expf(s[t][r] - mn); ps += s[t][r]; }
      ps += __shfl_xor(ps, 1);
      ps += __shfl_xor(ps, 2);
      ps += __shfl_xor(ps, 4);
      ps += __shfl_xor(ps, 8);
      lrow[r] = lrow[r] * alpha[r] + ps;
    }
#pragma unroll
    for (int oc = 0; oc < 8; oc++)
#pragma unroll
      for (int r = 0; r < 8; r++) acc[oc][r] *= alpha[r];

    // C-layout -> A-layout for P via per-wave LDS round trip
#pragma unroll
    for (int t = 0; t < 4; t++)
#pragma unroll
      for (int r = 0; r < 8; r++)
        sPw[(8 * lh + r) * 64 + t * 16 + n] = (__bf16)s[t][r];
    v16bf pa0 = load_frag_a(sPw, 64);
    v16bf pa1 = load_frag_a(sPw + 32, 64);
#pragma unroll
    for (int oc = 0; oc < 8; oc++) {
      v16bf bv0 = load_frag_b(sV + (oc * 16) * 64, 64);
      acc[oc] = wmma_bf16(pa0, bv0, acc[oc]);
      v16bf bv1 = load_frag_b(sV + (oc * 16) * 64 + 32, 64);
      acc[oc] = wmma_bf16(pa1, bv1, acc[oc]);
    }
  }

  // epilogue: out[b, head*128+d, l] = acc/lrow   (rows contiguous in l)
  float rv[8];
#pragma unroll
  for (int r = 0; r < 8; r++) rv[r] = 1.0f / lrow[r];
  const int bsel = bh >> 2, head = bh & 3;
  float* outp = out + ((size_t)bsel * 512 + head * 128) * L;
  const int lw = l0 + wv * 16 + 8 * lh;
#pragma unroll
  for (int oc = 0; oc < 8; oc++) {
    int d = oc * 16 + n;
    float4 lo, hi;
    lo.x = acc[oc][0] * rv[0]; lo.y = acc[oc][1] * rv[1];
    lo.z = acc[oc][2] * rv[2]; lo.w = acc[oc][3] * rv[3];
    hi.x = acc[oc][4] * rv[4]; hi.y = acc[oc][5] * rv[5];
    hi.z = acc[oc][6] * rv[6]; hi.w = acc[oc][7] * rv[7];
    *(float4*)(outp + (size_t)d * L + lw) = lo;
    *(float4*)(outp + (size_t)d * L + lw + 4) = hi;
  }
}

// ---------------------------------------------------------------------------
extern "C" void kernel_launch(void* const* d_in, const int* in_sizes, int n_in,
                              void* d_out, int out_size, void* d_ws,
                              size_t ws_size, hipStream_t stream) {
  (void)in_sizes; (void)n_in; (void)out_size; (void)ws_size;
  const float* fmap = (const float*)d_in[0];
  const float* w_qk = (const float*)d_in[1];
  const float* w_v  = (const float*)d_in[2];
  const float* relh = (const float*)d_in[3];
  const float* relw = (const float*)d_in[4];

  char* ws = (char*)d_ws;
  __bf16* Qb = (__bf16*)(ws);                      // 16 MiB bf16 [B,H,L,128]
  __bf16* Kb = (__bf16*)(ws + (16u << 20));        // 16 MiB bf16 [B,H,L,128]
  __bf16* Vt = (__bf16*)(ws + (32u << 20));        // 16 MiB bf16 [B,H,128,L]
  float* RWL = (float*)(ws + (48u << 20));         // 16 MiB f32 [B,H,L,64]
  float* RHL = (float*)(ws + (64u << 20));         // 16 MiB f32 [B,H,L,64]

  proj_kernel<<<dim3(8, 24, 16), 256, 0, stream>>>(fmap, w_qk, w_v, Qb, Kb, Vt);
  relproj_kernel<<<dim3(8, 64), 256, 0, stream>>>(Qb, relh, relw, RWL, RHL);
  attn_kernel<<<dim3(8, 64), 256, ATTN_LDS, stream>>>(Qb, Kb, Vt, RWL, RHL,
                                                      (float*)d_out);
}
